// SelfAttentionLayer_15204184228518
// MI455X (gfx1250) — matrix operand order repaired and checked by reading, not compiled, not run on hardware
//
#include <hip/hip_runtime.h>
#include <hip/hip_bf16.h>

// ---------------------------------------------------------------------------
// Self-attention layer for MI455X (gfx1250), wave32 + WMMA bf16 + async
// global->LDS copies (ASYNCcnt), two tiles in flight.
// Pipeline:
//   1. cast x, Wq, Wk, Wv, Wo fp32 -> bf16 (RNE)
//   2. Q/K/V = x @ W^T + b            (WMMA NT gemm, bf16 out)
//   3. per batch: S = Q K^T * 1/32    (WMMA NT gemm, fp32 out)
//                 P = softmax(S)      (row kernel, bf16 out)
//                 AO = P V            (WMMA NN gemm, bf16 out)
//   4. O = AO @ Wo^T + bo             (WMMA NT gemm, fp32 out)
//   5. y = LayerNorm(O + x) * gamma + beta
// ---------------------------------------------------------------------------

typedef __attribute__((ext_vector_type(16))) __bf16         v16bf;
typedef __attribute__((ext_vector_type(8)))  float          v8f;
typedef __attribute__((ext_vector_type(4)))  unsigned int   uint4v;
typedef __attribute__((ext_vector_type(16))) unsigned short ushort16;
typedef int int4g __attribute__((vector_size(16)));  // matches builtin param

struct Frag32B { uint4v lo; uint4v hi; };   // 32 bytes == v16bf

__device__ __forceinline__ v16bf make_frag(uint4v lo, uint4v hi) {
  Frag32B f{lo, hi};
  return __builtin_bit_cast(v16bf, f);
}

__device__ __forceinline__ unsigned short f32_to_bf16(float f) {
  unsigned int u = __builtin_bit_cast(unsigned int, f);
  u += 0x7fffu + ((u >> 16) & 1u);   // round-to-nearest-even
  return (unsigned short)(u >> 16);
}

// --- CDNA5 async global->LDS copy (GLOBAL_LOAD_ASYNC_TO_LDS_B128) ----------
#if defined(__AMDGCN__) &&                                          \
    __has_builtin(__builtin_amdgcn_global_load_async_to_lds_b128) && \
    __has_builtin(__builtin_amdgcn_s_wait_asynccnt)
#define USE_ASYNC_COPY 1
#else
#define USE_ASYNC_COPY 0
#endif

#define AS1 __attribute__((address_space(1)))
#define AS3 __attribute__((address_space(3)))
#if USE_ASYNC_COPY
#define ASYNC_CP16(gp, lp)                                  \
  __builtin_amdgcn_global_load_async_to_lds_b128(           \
      (AS1 int4g*)(gp), (AS3 int4g*)(lp), 0, 0)
#endif

// ---------------------------------------------------------------------------
// fp32 -> bf16 conversion (grid-stride)
// ---------------------------------------------------------------------------
__global__ __launch_bounds__(256) void cvt_f32_bf16(
    const float* __restrict__ src, unsigned short* __restrict__ dst, int n) {
  for (int i = blockIdx.x * blockDim.x + threadIdx.x; i < n;
       i += gridDim.x * blockDim.x)
    dst[i] = f32_to_bf16(src[i]);
}

// ---------------------------------------------------------------------------
// Tiled WMMA GEMM.  C[M,N] = scale * A[M,K] x Bop + bias
//   BT=true : B stored [N,K] row-major (C = A B^T)
//   BT=false: B stored [K,N] row-major (C = A B)
// Block tile 128x128x32, double-buffered LDS, async global->LDS staging with
// two tiles in flight (wait ASYNCcnt<=4 keeps the newest tile outstanding).
// 256 threads = 8 waves (wave32), wave grid 4x2, each wave 32x64 output
// (= 2x4 v_wmma_f32_16x16x32_bf16 accumulators).
// All problem dims are multiples of the tile sizes (no bounds checks).
// ---------------------------------------------------------------------------
#define BM  128
#define BN  128
#define BK  32
#define BKP 40    // padded K stride (halfs): 80B rows, 16B aligned, bank-skewed
#define BNP 136   // padded N stride for the NN B tile (272B rows)

template <bool BT, bool OUT_BF16>
__global__ __launch_bounds__(256) void gemm_wmma(
    const unsigned short* __restrict__ A, const unsigned short* __restrict__ B,
    const float* __restrict__ bias, void* __restrict__ Cout, int M, int N,
    int K, float scale) {
  constexpr int BROWS = BT ? BN : BK;
  constexpr int BCOLS = BT ? BKP : BNP;
  __shared__ alignas(16) unsigned short As[2][BM][BKP];
  __shared__ alignas(16) unsigned short Bs[2][BROWS][BCOLS];

  const int tid  = threadIdx.x;
  const int lane = tid & 31;
  const int wave = tid >> 5;
  const int wm   = wave >> 1;  // 0..3  -> M offset wm*32
  const int wn   = wave & 1;   // 0..1  -> N offset wn*64
  const int m0   = blockIdx.x * BM;
  const int n0   = blockIdx.y * BN;

  const int lhalf = lane >> 4;   // lane group 0/1
  const int l16   = lane & 15;
  const int akb   = lhalf * 16;  // A frag: k run  [akb,akb+16)
  const int bkb   = lhalf * 16;  // B frag: k run  [bkb,bkb+16)

  // per-thread staging coordinates (16B chunks, 2 per tensor per thread)
  const int ar0 = tid >> 2;          // A/B(NT) tile row, chunk0 (0..63)
  const int as8 = (tid & 3) * 8;     // half offset within row
  const int bkr = tid >> 4;          // NN B tile k-row, chunk0 (0..15)
  const int bs8 = (tid & 15) * 8;    // NN half offset within row

  const unsigned short* gA0 = A + (size_t)(m0 + ar0) * K + as8;
  const unsigned short* gA1 = gA0 + (size_t)64 * K;
  const unsigned short* gB0;
  const unsigned short* gB1;
  if constexpr (BT) {
    gB0 = B + (size_t)(n0 + ar0) * K + as8;
    gB1 = gB0 + (size_t)64 * K;
  } else {
    gB0 = B + (size_t)bkr * N + n0 + bs8;
    gB1 = gB0 + (size_t)16 * N;
  }
  const size_t bstep = BT ? (size_t)BK : (size_t)BK * N;

  v8f acc[2][4];
  const v8f vzero = {0.f, 0.f, 0.f, 0.f, 0.f, 0.f, 0.f, 0.f};
#pragma unroll
  for (int i = 0; i < 2; i++)
#pragma unroll
    for (int j = 0; j < 4; j++) acc[i][j] = vzero;

  v16bf afrag[2], bfrag[4];
  auto read_frags = [&](int cur) {
#pragma unroll
    for (int mi = 0; mi < 2; mi++) {
      int r = wm * 32 + mi * 16 + l16;
      afrag[mi] = make_frag(*(const uint4v*)&As[cur][r][akb],
                            *(const uint4v*)&As[cur][r][akb + 8]);
    }
#pragma unroll
    for (int ni = 0; ni < 4; ni++) {
      if constexpr (BT) {
        int r = wn * 64 + ni * 16 + l16;
        bfrag[ni] = make_frag(*(const uint4v*)&Bs[cur][r][bkb],
                              *(const uint4v*)&Bs[cur][r][bkb + 8]);
      } else {
        int ncol = wn * 64 + ni * 16 + l16;
        ushort16 bu;
#pragma unroll
        for (int j = 0; j < 16; j++) bu[j] = Bs[cur][bkb + j][ncol];
        bfrag[ni] = __builtin_bit_cast(v16bf, bu);
      }
    }
  };
  auto do_wmma = [&]() {
#pragma unroll
    for (int mi = 0; mi < 2; mi++)
#pragma unroll
      for (int ni = 0; ni < 4; ni++)
        acc[mi][ni] = __builtin_amdgcn_wmma_f32_16x16x32_bf16(
            false, afrag[mi], false, bfrag[ni], (short)0, acc[mi][ni], false,
            false);
  };
  auto advance = [&]() {
    gA0 += BK; gA1 += BK; gB0 += bstep; gB1 += bstep;
  };

  const int KT = K / BK;

#if USE_ASYNC_COPY
  // -------- async pipeline, two tiles in flight (ASYNCcnt) -----------------
  auto issue_stage = [&](int buf) {
    ASYNC_CP16(gA0, &As[buf][ar0][as8]);
    ASYNC_CP16(gA1, &As[buf][ar0 + 64][as8]);
    if constexpr (BT) {
      ASYNC_CP16(gB0, &Bs[buf][ar0][as8]);
      ASYNC_CP16(gB1, &Bs[buf][ar0 + 64][as8]);
    } else {
      ASYNC_CP16(gB0, &Bs[buf][bkr][bs8]);
      ASYNC_CP16(gB1, &Bs[buf][bkr + 16][bs8]);
    }
  };
  issue_stage(0);
  advance();
  for (int kt = 0; kt < KT; ++kt) {
    const int cur = kt & 1;
    if (kt + 1 < KT) {
      // writes to buf cur^1: every thread's reads of cur^1 finished before
      // the second barrier of iteration kt-1, which precedes this point.
      issue_stage(cur ^ 1);
      advance();
      // tile kt's 4 copies (older, in-order) done; tile kt+1 stays in flight
      __builtin_amdgcn_s_wait_asynccnt(4);
    } else {
      __builtin_amdgcn_s_wait_asynccnt(0);
    }
    __syncthreads();     // tile kt visible to all threads
    read_frags(cur);
    __syncthreads();     // all reads of buf cur done -> may be overwritten
    do_wmma();
  }
#else
  // -------- fallback: register-staged double-buffered pipeline -------------
  uint4v rA0, rA1, rB0, rB1;
  auto load_regs = [&]() {
    rA0 = *(const uint4v*)gA0;
    rA1 = *(const uint4v*)gA1;
    rB0 = *(const uint4v*)gB0;
    rB1 = *(const uint4v*)gB1;
  };
  auto store_regs = [&](int buf) {
    *(uint4v*)&As[buf][ar0][as8]      = rA0;
    *(uint4v*)&As[buf][ar0 + 64][as8] = rA1;
    if constexpr (BT) {
      *(uint4v*)&Bs[buf][ar0][as8]      = rB0;
      *(uint4v*)&Bs[buf][ar0 + 64][as8] = rB1;
    } else {
      *(uint4v*)&Bs[buf][bkr][bs8]      = rB0;
      *(uint4v*)&Bs[buf][bkr + 16][bs8] = rB1;
    }
  };
  load_regs();
  advance();
  for (int kt = 0; kt < KT; ++kt) {
    const int cur = kt & 1;
    store_regs(cur);
    __syncthreads();
    read_frags(cur);
    if (kt + 1 < KT) { load_regs(); advance(); }
    do_wmma();
  }
#endif

  // ---- epilogue: scale, bias, store (C layout: VGPR r -> row r + 8*lhalf) --
#pragma unroll
  for (int mi = 0; mi < 2; mi++) {
#pragma unroll
    for (int ni = 0; ni < 4; ni++) {
      int colg = n0 + wn * 64 + ni * 16 + l16;
      float bv = bias ? bias[colg] : 0.f;
#pragma unroll
      for (int r = 0; r < 8; r++) {
        int rowg = m0 + wm * 32 + mi * 16 + lhalf * 8 + r;
        float v = acc[mi][ni][r] * scale + bv;
        if constexpr (OUT_BF16)
          ((unsigned short*)Cout)[(size_t)rowg * N + colg] = f32_to_bf16(v);
        else
          ((float*)Cout)[(size_t)rowg * N + colg] = v;
      }
    }
  }
}

// ---------------------------------------------------------------------------
// Row softmax: one block (256 threads) per row, fp32 in -> bf16 out
// ---------------------------------------------------------------------------
__global__ __launch_bounds__(256) void softmax_rows(
    const float* __restrict__ S, unsigned short* __restrict__ P, int cols) {
  const int row = blockIdx.x;
  const float* s = S + (size_t)row * cols;
  unsigned short* p = P + (size_t)row * cols;
  __shared__ float red[256];
  const int tid = threadIdx.x;

  float m = -3.4e38f;
  for (int j = tid; j < cols; j += 256) m = fmaxf(m, s[j]);
  red[tid] = m;
  __syncthreads();
  for (int off = 128; off > 0; off >>= 1) {
    if (tid < off) red[tid] = fmaxf(red[tid], red[tid + off]);
    __syncthreads();
  }
  m = red[0];
  __syncthreads();

  float sum = 0.f;
  for (int j = tid; j < cols; j += 256) sum += __expf(s[j] - m);
  red[tid] = sum;
  __syncthreads();
  for (int off = 128; off > 0; off >>= 1) {
    if (tid < off) red[tid] += red[tid + off];
    __syncthreads();
  }
  float inv = 1.f / red[0];

  for (int j = tid; j < cols; j += 256)
    p[j] = f32_to_bf16(__expf(s[j] - m) * inv);
}

// ---------------------------------------------------------------------------
// y = LayerNorm(proj + x) * gamma + beta  (one block per row)
// ---------------------------------------------------------------------------
__global__ __launch_bounds__(256) void ln_residual(
    const float* __restrict__ proj, const float* __restrict__ x,
    const float* __restrict__ gamma, const float* __restrict__ beta,
    float* __restrict__ out, int cols) {
  const int row = blockIdx.x;
  const int tid = threadIdx.x;
  const float* pr = proj + (size_t)row * cols;
  const float* xr = x + (size_t)row * cols;
  float* orow = out + (size_t)row * cols;
  __shared__ float red[256];

  float s = 0.f;
  for (int j = tid; j < cols; j += 256) s += pr[j] + xr[j];
  red[tid] = s;
  __syncthreads();
  for (int off = 128; off > 0; off >>= 1) {
    if (tid < off) red[tid] += red[tid + off];
    __syncthreads();
  }
  float mu = red[0] / cols;
  __syncthreads();

  float vs = 0.f;
  for (int j = tid; j < cols; j += 256) {
    float d = pr[j] + xr[j] - mu;
    vs += d * d;
  }
  red[tid] = vs;
  __syncthreads();
  for (int off = 128; off > 0; off >>= 1) {
    if (tid < off) red[tid] += red[tid + off];
    __syncthreads();
  }
  float rstd = rsqrtf(red[0] / cols + 1e-5f);

  for (int j = tid; j < cols; j += 256) {
    float d = pr[j] + xr[j] - mu;
    orow[j] = d * rstd * gamma[j] + beta[j];
  }
}

// ---------------------------------------------------------------------------
// launcher
// ---------------------------------------------------------------------------
extern "C" void kernel_launch(void* const* d_in, const int* in_sizes, int n_in,
                              void* d_out, int out_size, void* d_ws,
                              size_t ws_size, hipStream_t stream) {
  (void)in_sizes; (void)n_in; (void)out_size; (void)ws_size;
  constexpr int    Bb = 8, S = 2048, D = 1024, H = 1024;
  constexpr size_t MS = (size_t)Bb * S;  // 16384 total rows

  const float* x  = (const float*)d_in[0];
  const float* Wq = (const float*)d_in[1];
  const float* bq = (const float*)d_in[2];
  const float* Wk = (const float*)d_in[3];
  const float* bk = (const float*)d_in[4];
  const float* Wv = (const float*)d_in[5];
  const float* bv = (const float*)d_in[6];
  const float* Wo = (const float*)d_in[7];
  const float* bo = (const float*)d_in[8];
  const float* gamma = (const float*)d_in[9];
  const float* beta  = (const float*)d_in[10];

  // workspace carve-up (~256 MB total)
  char* ws = (char*)d_ws;
  size_t off = 0;
  auto carve = [&](size_t bytes) -> char* {
    char* p = ws + off;
    off = (off + bytes + 255) & ~(size_t)255;
    return p;
  };
  unsigned short* xb  = (unsigned short*)carve(MS * D * 2);        // 32 MB
  unsigned short* Wqb = (unsigned short*)carve((size_t)H * D * 2); //  2 MB
  unsigned short* Wkb = (unsigned short*)carve((size_t)H * D * 2);
  unsigned short* Wvb = (unsigned short*)carve((size_t)H * D * 2);
  unsigned short* Wob = (unsigned short*)carve((size_t)H * H * 2);
  unsigned short* Qb  = (unsigned short*)carve(MS * H * 2);        // 32 MB
  unsigned short* Kb  = (unsigned short*)carve(MS * H * 2);
  unsigned short* Vb  = (unsigned short*)carve(MS * H * 2);
  float*          Sf  = (float*)carve((size_t)S * S * 4);          // 16 MB
  unsigned short* Pb  = (unsigned short*)carve((size_t)S * S * 2); //  8 MB
  unsigned short* AOb = (unsigned short*)carve(MS * H * 2);        // 32 MB
  float*          Of  = (float*)carve(MS * H * 4);                 // 64 MB

  const dim3 blk(256);

  // 1. cast inputs to bf16
  cvt_f32_bf16<<<2048, blk, 0, stream>>>(x, xb, (int)(MS * D));
  cvt_f32_bf16<<<512, blk, 0, stream>>>(Wq, Wqb, H * D);
  cvt_f32_bf16<<<512, blk, 0, stream>>>(Wk, Wkb, H * D);
  cvt_f32_bf16<<<512, blk, 0, stream>>>(Wv, Wvb, H * D);
  cvt_f32_bf16<<<512, blk, 0, stream>>>(Wo, Wob, H * H);

  // 2. Q/K/V projections: [16384,1024] x [1024,1024]^T
  const dim3 gQKV(MS / BM, H / BN);
  gemm_wmma<true, true><<<gQKV, blk, 0, stream>>>(xb, Wqb, bq, Qb, (int)MS, H, D, 1.f);
  gemm_wmma<true, true><<<gQKV, blk, 0, stream>>>(xb, Wkb, bk, Kb, (int)MS, H, D, 1.f);
  gemm_wmma<true, true><<<gQKV, blk, 0, stream>>>(xb, Wvb, bv, Vb, (int)MS, H, D, 1.f);

  // 3. attention, per batch (scores buffer reused)
  const float scale = 0.03125f;  // 1/sqrt(1024)
  for (int b = 0; b < Bb; b++) {
    const unsigned short* Qx = Qb + (size_t)b * S * H;
    const unsigned short* Kx = Kb + (size_t)b * S * H;
    const unsigned short* Vx = Vb + (size_t)b * S * H;
    unsigned short* AOx = AOb + (size_t)b * S * H;
    gemm_wmma<true, false><<<dim3(S / BM, S / BN), blk, 0, stream>>>(
        Qx, Kx, nullptr, Sf, S, S, H, scale);
    softmax_rows<<<S, blk, 0, stream>>>(Sf, Pb, S);
    gemm_wmma<false, true><<<dim3(S / BM, H / BN), blk, 0, stream>>>(
        Pb, Vx, nullptr, AOx, S, H, S, 1.f);
  }

  // 4. output projection
  gemm_wmma<true, false><<<dim3(MS / BM, H / BN), blk, 0, stream>>>(
      AOb, Wob, bo, Of, (int)MS, H, H, 1.f);

  // 5. residual + LayerNorm -> d_out
  ln_residual<<<MS, blk, 0, stream>>>(Of, x, gamma, beta, (float*)d_out, H);
}
